// STGCN_12695923327317
// MI455X (gfx1250) — matrix operand order repaired and checked by reading, not compiled, not run on hardware
//
#include <hip/hip_runtime.h>
#include <hip/hip_bf16.h>

// ---------------- problem constants ----------------
#define BB 32
#define NN 2048
#define CC 64     // CIN
#define OO 128    // COUT
#define MT 128    // m-tile (columns of A) per workgroup
#define XS_STR 40 // halves per row of xs chunk in LDS (32 used + pad, 80B: 16B aligned, conflict-free)
#define W_STR  72 // halves per row of lds_w / lds_xg (64 used + pad, 144B: 16B aligned, conflict-free)

typedef __attribute__((ext_vector_type(16))) __bf16 bf16x16;
typedef __attribute__((ext_vector_type(8)))  float  f32x8;
typedef __attribute__((ext_vector_type(4)))  int    v4i;

typedef __attribute__((address_space(1))) v4i* gv4i_p; // global
typedef __attribute__((address_space(3))) v4i* lv4i_p; // LDS

union BV {
    bf16x16 v;
    uint4 q[2];
    unsigned short h[16];
};

// f32 -> bf16 via native convert (v_cvt_pk_bf16_f32 on gfx1250), RNE
__device__ __forceinline__ unsigned short f2bf(float f) {
    union { __bf16 h; unsigned short u; } cv;
    cv.h = (__bf16)f;
    return cv.u;
}

// CDNA5 async global->LDS copy support
#if defined(__gfx1250__) && __has_builtin(__builtin_amdgcn_global_load_async_to_lds_b128)
#define HAVE_ASYNC 1
#else
#define HAVE_ASYNC 0
#endif

#if __has_builtin(__builtin_amdgcn_s_wait_asynccnt)
#define WAIT_ASYNC0() __builtin_amdgcn_s_wait_asynccnt(0)
#else
#define WAIT_ASYNC0() asm volatile("s_wait_asynccnt 0" ::: "memory")
#endif

// ---------------- K1: row sums of A -> 1/(sum+eps) ----------------
__global__ void k_rowsum(const float* __restrict__ A, float* __restrict__ rinv) {
    __shared__ float red[256];
    const int row = blockIdx.x; // b*NN + n
    const float* ap = A + (size_t)row * NN;
    float s = 0.f;
    for (int i = threadIdx.x; i < NN; i += 256) s += ap[i];
    red[threadIdx.x] = s;
    __syncthreads();
    for (int off = 128; off > 0; off >>= 1) {
        if (threadIdx.x < (unsigned)off) red[threadIdx.x] += red[threadIdx.x + off];
        __syncthreads();
    }
    if (threadIdx.x == 0) rinv[row] = 1.0f / (red[0] + 1e-6f);
}

// ---------------- K2: xs_t[b][c][n] = bf16(x[b][n][c] * rinv[b,n]) ----------------
__global__ void k_xs(const float* __restrict__ x, const float* __restrict__ rinv,
                     unsigned short* __restrict__ xs_t) {
    __shared__ __align__(16) unsigned short tile[CC][XS_STR]; // [c][n_local]
    const int b  = blockIdx.x >> 6;           // 64 n-tiles of 32 per batch
    const int n0 = (blockIdx.x & 63) * 32;
    const int t  = threadIdx.x;
    {
        const int nl = t >> 3;                // 0..31
        const int c8 = (t & 7) * 8;           // 0..56
        const float* xp = x + ((size_t)(b * NN + n0 + nl)) * CC + c8;
        const float r = rinv[b * NN + n0 + nl];
        float4 v0 = *(const float4*)(xp);
        float4 v1 = *(const float4*)(xp + 4);
        tile[c8 + 0][nl] = f2bf(v0.x * r);
        tile[c8 + 1][nl] = f2bf(v0.y * r);
        tile[c8 + 2][nl] = f2bf(v0.z * r);
        tile[c8 + 3][nl] = f2bf(v0.w * r);
        tile[c8 + 4][nl] = f2bf(v1.x * r);
        tile[c8 + 5][nl] = f2bf(v1.y * r);
        tile[c8 + 6][nl] = f2bf(v1.z * r);
        tile[c8 + 7][nl] = f2bf(v1.w * r);
    }
    __syncthreads();
    {
        const int c = t >> 2, seg = t & 3;
        uint4 v = *(const uint4*)(&tile[c][seg * 8]);
        *(uint4*)(xs_t + (size_t)(b * CC + c) * NN + n0 + seg * 8) = v;
    }
}

// ---------------- K3: fused GEMM1 (xs^T * A) + GEMM2 (W mix) + bias + BN stats ----------------
__global__ __launch_bounds__(256)
void k_main(const float* __restrict__ A, const unsigned short* __restrict__ xs_t,
            const float* __restrict__ W, const float* __restrict__ bias,
            float* __restrict__ out, float* __restrict__ stats) {
    __shared__ __align__(16) unsigned short lds_w[OO * W_STR];       // W as bf16 [o][c]
    __shared__ __align__(16) unsigned short lds_xs[2][CC * XS_STR];  // double-buffered xs chunk [c][k]
    __shared__ __align__(16) unsigned short lds_xg[MT * W_STR];      // xg^T as bf16 [m][c]
    __shared__ float lds_stats[2 * OO];

    const int b    = blockIdx.x >> 4;
    const int m0   = (blockIdx.x & 15) * MT;
    const int tid  = threadIdx.x;
    const int wave = tid >> 5;
    const int lane = tid & 31;
    const int l15  = lane & 15;
    const bool hi  = lane >= 16;

    // Stage W -> LDS as bf16 (padded rows)
    for (int i = tid; i < OO * CC; i += 256) {
        const int o = i >> 6, c = i & 63;
        lds_w[o * W_STR + c] = f2bf(W[i]);
    }
    if (tid < 2 * OO) lds_stats[tid] = 0.f;

    const unsigned short* xsg = xs_t + (size_t)b * CC * NN;
    // Stage one 32xCC xs chunk into LDS; async DMA on CDNA5 (no VGPR round-trip)
    auto stage = [&](int buf, int n0c) {
        const int c = tid >> 2, seg = tid & 3;
        const unsigned short* src = xsg + (size_t)c * NN + n0c + seg * 8;
        unsigned short* dst = &lds_xs[buf][c * XS_STR + seg * 8];
#if HAVE_ASYNC
        __builtin_amdgcn_global_load_async_to_lds_b128(
            (gv4i_p)src, (lv4i_p)dst, 0, 0);
#else
        *(uint4*)dst = *(const uint4*)src;
#endif
    };
    stage(0, 0);

    // -------- stage 1: xg[c, m] = sum_n xs[n,c] * A[n,m], contraction chunks of 32 --------
    f32x8 acc[4] = {};  // 4 c-tiles of 16 x this wave's 16 m-columns

    const int colm = m0 + wave * 16 + l15;      // this lane's m column
    const int k0   = hi ? 8 : 0;                // B-operand K offset per ISA layout
    const int klo  = hi ? 8 : 0;                // A-operand K offset per ISA layout
    const float* Ab = A + (size_t)b * NN * NN + (size_t)k0 * NN + colm;

    for (int ch = 0; ch < 64; ++ch) {
        const int n0 = ch * 32;

        // Stream A directly from HBM (each element used exactly once -> no LDS for A)
        float av[16];
#pragma unroll
        for (int r = 0; r < 16; ++r) av[r] = Ab[(size_t)(n0 + r + (r >= 8 ? 8 : 0)) * NN];
        if (ch < 63) __builtin_prefetch(&Ab[(size_t)(n0 + 32) * NN], 0, 1);

#if HAVE_ASYNC
        WAIT_ASYNC0();                   // our async stores into lds_xs landed
#endif
        __syncthreads();                 // current xs buffer ready; prev buffer reads done
        if (ch < 63) stage((ch + 1) & 1, n0 + 32);

        // Pack A chunk to bf16 in B-operand layout (lane holds K = k0 + 0..7, k0 + 16..23)
        // native casts -> v_cvt_pk_bf16_f32
        BV bv;
#pragma unroll
        for (int r = 0; r < 16; ++r) bv.v[r] = (__bf16)av[r];

        const unsigned short* xsbuf = lds_xs[ch & 1];
#pragma unroll
        for (int ct = 0; ct < 4; ++ct) {
            const unsigned short* p = xsbuf + (ct * 16 + l15) * XS_STR + klo;
            BV avv;
            avv.q[0] = *(const uint4*)(p);        // K = klo .. klo+7
            avv.q[1] = *(const uint4*)(p + 16);   // K = klo+16 .. klo+23
            acc[ct] = __builtin_amdgcn_wmma_f32_16x16x32_bf16(
                false, avv.v, false, bv.v, (short)0, acc[ct], false, false);
        }
    }

    // -------- write xg accumulators to LDS as bf16 in [m][c] layout --------
    __syncthreads();
    {
        const int mrow = wave * 16 + l15;       // D: N = m = l15 within wave tile
#pragma unroll
        for (int ct = 0; ct < 4; ++ct) {
#pragma unroll
            for (int r = 0; r < 8; r += 2) {    // D: M = c = ct*16 + r + 8*hi
                const int c = ct * 16 + r + (hi ? 8 : 0);
                union { __bf16 h2[2]; unsigned u; } pk;
                pk.h2[0] = (__bf16)acc[ct][r];
                pk.h2[1] = (__bf16)acc[ct][r + 1];
                *(unsigned*)(&lds_xg[mrow * W_STR + c]) = pk.u;
            }
        }
    }
    __syncthreads();

    // -------- stage 2: h^T[m, o] = sum_c xg[c,m] * W[o,c] + bias[o] --------
    BV a2[2];   // xg^T block for this wave's 16 m rows, 2 c-chunks of 32 (reused over 8 o-tiles)
    {
        const int mrow = wave * 16 + l15;
#pragma unroll
        for (int cc2 = 0; cc2 < 2; ++cc2) {
            const unsigned short* p = &lds_xg[mrow * W_STR + cc2 * 32 + klo];
            a2[cc2].q[0] = *(const uint4*)(p);
            a2[cc2].q[1] = *(const uint4*)(p + 16);
        }
    }
    const int koffB = hi ? 8 : 0;
    float* outp = out + ((size_t)b * NN + m0) * OO;

#pragma unroll 1
    for (int ot = 0; ot < 8; ++ot) {
        const int o = ot * 16 + l15;
        const float bb = bias[o];
        f32x8 h;
#pragma unroll
        for (int r = 0; r < 8; ++r) h[r] = bb;   // bias folded into accumulator

#pragma unroll
        for (int cc2 = 0; cc2 < 2; ++cc2) {
            const unsigned short* p = &lds_w[o * W_STR + cc2 * 32 + koffB];
            BV bw;
            bw.q[0] = *(const uint4*)(p);
            bw.q[1] = *(const uint4*)(p + 16);
            h = __builtin_amdgcn_wmma_f32_16x16x32_bf16(
                false, a2[cc2].v, false, bw.v, (short)0, h, false, false);
        }

        // store [m, o] tile directly into out[b][m][o] (reference's final transpose for free)
        float s1 = 0.f, s2 = 0.f;
#pragma unroll
        for (int r = 0; r < 8; ++r) {
            const int m = wave * 16 + r + (hi ? 8 : 0);
            outp[(size_t)m * OO + o] = h[r];
            s1 += h[r];
            s2 += h[r] * h[r];
        }
        // lanes l and l+16 hold the same o -> combine halves, then LDS-atomic per channel
        s1 += __shfl_xor(s1, 16, 32);
        s2 += __shfl_xor(s2, 16, 32);
        if (!hi) {
            atomicAdd(&lds_stats[o], s1);
            atomicAdd(&lds_stats[OO + o], s2);
        }
    }
    __syncthreads();
    if (tid < 2 * OO) atomicAdd(&stats[tid], lds_stats[tid]);
}

// ---------------- K4: BatchNorm (train-mode batch stats) + ReLU, in place ----------------
__global__ void k_bnrelu(float* __restrict__ out, const float* __restrict__ stats,
                         const float* __restrict__ gamma, const float* __restrict__ beta) {
    __shared__ float sc[OO], sh[OO];
    if (threadIdx.x < OO) {
        const float cnt  = (float)(BB * NN);
        const float mean = stats[threadIdx.x] / cnt;
        const float var  = stats[OO + threadIdx.x] / cnt - mean * mean;
        const float s    = gamma[threadIdx.x] * rsqrtf(var + 1e-5f);
        sc[threadIdx.x] = s;
        sh[threadIdx.x] = beta[threadIdx.x] - mean * s;
    }
    __syncthreads();
    const size_t base = (size_t)blockIdx.x * 4096;
#pragma unroll
    for (int i = 0; i < 4; ++i) {
        const size_t idx = base + (size_t)i * 1024 + (size_t)threadIdx.x * 4;
        float4 v = *(float4*)(out + idx);
        const int o = (int)(idx & 127);
        v.x = fmaxf(v.x * sc[o + 0] + sh[o + 0], 0.f);
        v.y = fmaxf(v.y * sc[o + 1] + sh[o + 1], 0.f);
        v.z = fmaxf(v.z * sc[o + 2] + sh[o + 2], 0.f);
        v.w = fmaxf(v.w * sc[o + 3] + sh[o + 3], 0.f);
        *(float4*)(out + idx) = v;
    }
}

// ---------------- host launcher ----------------
extern "C" void kernel_launch(void* const* d_in, const int* in_sizes, int n_in,
                              void* d_out, int out_size, void* d_ws, size_t ws_size,
                              hipStream_t stream) {
    const float* x     = (const float*)d_in[0];
    const float* A     = (const float*)d_in[1];
    const float* W     = (const float*)d_in[2];
    const float* bias  = (const float*)d_in[3];
    const float* gamma = (const float*)d_in[4];
    const float* beta  = (const float*)d_in[5];
    float* out = (float*)d_out;

    char* ws = (char*)d_ws;
    float* rinv           = (float*)ws;                                   // 256 KB
    unsigned short* xs_t  = (unsigned short*)(ws + (size_t)BB * NN * 4);  // 8 MB bf16 [b][c][n]
    float* stats          = (float*)(ws + (size_t)BB * NN * 4 +
                                          (size_t)BB * CC * NN * 2);      // 2*128 floats

    (void)hipMemsetAsync(stats, 0, 2 * OO * sizeof(float), stream);
    k_rowsum<<<BB * NN, 256, 0, stream>>>(A, rinv);
    k_xs<<<BB * (NN / 32), 256, 0, stream>>>(x, rinv, xs_t);
    k_main<<<BB * (NN / MT), 256, 0, stream>>>(A, xs_t, W, bias, out, stats);
    k_bnrelu<<<(BB * NN * OO) / 4096, 256, 0, stream>>>(out, stats, gamma, beta);
}
